// LanguageMoE_28063316312422
// MI455X (gfx1250) — matrix operand order repaired and checked by reading, not compiled, not run on hardware
//
#include <hip/hip_runtime.h>
#include <hip/hip_bf16.h>

// ---------------------------------------------------------------------------
// LanguageMoE on gfx1250: all GEMMs via v_wmma_f32_16x16x32_f16 (f16 in, f32 acc)
// + GLOBAL_LOAD_ASYNC_TO_LDS_B128 staging in attention, NT stores for logits.
// ---------------------------------------------------------------------------

#define B_      32
#define N_      125
#define D_      256
#define H_      8
#define HD_     32
#define FF_     1024
#define E_      5
#define V_      32000
#define HID_    128
#define S_      128          // N_ + 3
#define MAXPOS_ 512
#define EPS_    1e-12f
#define THRESH_ 0.7f

typedef __attribute__((ext_vector_type(16))) _Float16 v16h;
typedef __attribute__((ext_vector_type(8)))  _Float16 v8h;
typedef __attribute__((ext_vector_type(8)))  float    v8f;
typedef _Float16 h16;

static __device__ __forceinline__ v16h cat8(v8h lo, v8h hp) {
    return __builtin_shufflevector(lo, hp, 0,1,2,3,4,5,6,7,8,9,10,11,12,13,14,15);
}

// ---------------------------------------------------------------------------
// Generic WMMA GEMM: C[M,N] = A[M,K](f16) * Bt[N,K](f16, pre-transposed) + bias
// 256 threads = 8 waves; block tile 128x128; wave tile 32(M)x64(N); K step 32.
// act: 0 = none, 1 = exact GELU. nt: nontemporal f32 stores (streaming output).
// Output either f32 (Cf) or f16 (Ch).
// Requires M%128==0, N%128==0, K%32==0 (all shapes here satisfy this).
// ---------------------------------------------------------------------------
__global__ __launch_bounds__(256) void gemm_wmma_kernel(
    const h16* __restrict__ A, const h16* __restrict__ Bt,
    const float* __restrict__ bias,
    float* __restrict__ Cf, h16* __restrict__ Ch,
    int M, int N, int K, int act, int nt)
{
    const int lane = threadIdx.x & 31;
    const int wave = threadIdx.x >> 5;
    const int wm = wave >> 1, wn = wave & 1;
    const int l  = lane & 15, hi = lane >> 4;
    const int rowBase = blockIdx.y * 128 + wm * 32;
    const int colBase = blockIdx.x * 128 + wn * 64;

    const v8f zero = {0.f,0.f,0.f,0.f,0.f,0.f,0.f,0.f};
    v8f acc[2][4];
#pragma unroll
    for (int i = 0; i < 2; ++i)
#pragma unroll
        for (int j = 0; j < 4; ++j) acc[i][j] = zero;

    for (int k0 = 0; k0 < K; k0 += 32) {
        // prefetch next K tile (global_prefetch_b8); whole cacheline per lane
        if (k0 + 32 < K) {
            __builtin_prefetch(A  + (size_t)(rowBase + 16*(lane & 1) + l) * K + k0 + 32, 0, 3);
            __builtin_prefetch(Bt + (size_t)(colBase + 16*(lane & 3) + l) * K + k0 + 32, 0, 3);
        }
        v16h a[2], b[4];
#pragma unroll
        for (int i = 0; i < 2; ++i) {
            const h16* ap = A + (size_t)(rowBase + 16*i + l) * K + k0 + hi*8;
            v8h lo = *(const v8h*)ap;
            v8h hp = *(const v8h*)(ap + 16);
            a[i] = cat8(lo, hp);
        }
#pragma unroll
        for (int j = 0; j < 4; ++j) {
            const h16* bp = Bt + (size_t)(colBase + 16*j + l) * K + k0 + hi*8;
            b[j] = cat8(*(const v8h*)bp, *(const v8h*)(bp + 16));
        }
#pragma unroll
        for (int i = 0; i < 2; ++i)
#pragma unroll
            for (int j = 0; j < 4; ++j)
                acc[i][j] = __builtin_amdgcn_wmma_f32_16x16x32_f16(
                    false, a[i], false, b[j], (short)0, acc[i][j], false, false);
    }

#pragma unroll
    for (int i = 0; i < 2; ++i)
#pragma unroll
        for (int j = 0; j < 4; ++j) {
            const int col = colBase + 16*j + l;
            const float bv = bias ? bias[col] : 0.f;
#pragma unroll
            for (int r = 0; r < 8; ++r) {
                const int row = rowBase + 16*i + hi*8 + r;
                float v = acc[i][j][r] + bv;
                if (act == 1) v = 0.5f * v * (1.f + erff(v * 0.70710678118654752f));
                const size_t o = (size_t)row * N + col;
                if (Ch)      Ch[o] = (h16)v;
                else if (nt) __builtin_nontemporal_store(v, &Cf[o]);
                else         Cf[o] = v;
            }
        }
}

// ---------------------------------------------------------------------------
// Attention: one block per (b,h). q,k,v f16 [B*S, D]; ctx f16 [B*S, D].
// q/k staged into LDS via GLOBAL_LOAD_ASYNC_TO_LDS_B128 (ASYNCcnt),
// v transposed on the scalar path (overlaps with in-flight async copies).
// scores = q@k^T * 1/sqrt(HD), softmax, ctx = P @ v. All matmuls via WMMA.
// ---------------------------------------------------------------------------
struct AttnSmem {
    alignas(32) h16 qs[S_ * HD_];   // [s][d]
    alignas(32) h16 ks[S_ * HD_];   // [s][d]
    alignas(32) h16 vsT[HD_ * S_];  // [d][s]  (transposed)
    alignas(32) h16 sc[S_ * S_];    // scores then probs, f16
};

__global__ __launch_bounds__(256) void attn_kernel(
    const h16* __restrict__ qg, const h16* __restrict__ kg,
    const h16* __restrict__ vg, h16* __restrict__ ctx)
{
    __shared__ AttnSmem sm;
    const int bIdx = blockIdx.x >> 3;
    const int hIdx = blockIdx.x & 7;
    const int t = threadIdx.x;

    // async global->LDS copies of q and k: 16B per lane per issue
#pragma unroll
    for (int it = 0; it < 2; ++it) {
        const int idx = t + it * 256;              // 512 16B-chunks per matrix
        const int s = idx >> 2, c4 = idx & 3;
        const size_t g = ((size_t)(bIdx * S_ + s)) * D_ + hIdx * HD_ + c4 * 8;
        const unsigned qOff = (unsigned)(size_t)&sm.qs[s * HD_ + c4 * 8];
        const unsigned kOff = (unsigned)(size_t)&sm.ks[s * HD_ + c4 * 8];
        const unsigned long long qa = (unsigned long long)(const void*)(qg + g);
        const unsigned long long ka = (unsigned long long)(const void*)(kg + g);
        asm volatile("global_load_async_to_lds_b128 %0, %1, off"
                     :: "v"(qOff), "v"(qa) : "memory");
        asm volatile("global_load_async_to_lds_b128 %0, %1, off"
                     :: "v"(kOff), "v"(ka) : "memory");
    }
    // v: transpose while async copies are in flight
    for (int idx = t; idx < S_ * HD_; idx += 256) {
        const int s = idx >> 5, c = idx & 31;
        sm.vsT[c * S_ + s] = vg[((size_t)(bIdx * S_ + s)) * D_ + hIdx * HD_ + c];
    }
    asm volatile("s_wait_asynccnt 0x0" ::: "memory");
    __syncthreads();

    const int lane = t & 31, wave = t >> 5;
    const int l = lane & 15, hi = lane >> 4;
    const v8f zero = {0.f,0.f,0.f,0.f,0.f,0.f,0.f,0.f};

    // scores: wave handles rows 16*wave..16*wave+15 across all 128 cols
    {
        const h16* ap = sm.qs + (16*wave + l) * HD_ + hi*8;
        v8h lo = *(const v8h*)ap;
        v8h hp = *(const v8h*)(ap + 16);
        v16h a = cat8(lo, hp);
#pragma unroll
        for (int j = 0; j < 8; ++j) {
            const h16* bp = sm.ks + (16*j + l) * HD_ + hi*8;
            v16h bb = cat8(*(const v8h*)bp, *(const v8h*)(bp + 16));
            v8f c = zero;
            c = __builtin_amdgcn_wmma_f32_16x16x32_f16(false, a, false, bb, (short)0, c, false, false);
#pragma unroll
            for (int r = 0; r < 8; ++r)
                sm.sc[(16*wave + hi*8 + r) * S_ + 16*j + l] = (h16)c[r];
        }
    }
    __syncthreads();

    // row softmax (scale applied here), probs written back in place as f16
    if (t < S_) {
        const float SCALE = 0.17677669529663687f;   // 1/sqrt(32)
        h16* row = sm.sc + t * S_;
        float m = -1e30f;
        for (int c = 0; c < S_; ++c) m = fmaxf(m, (float)row[c] * SCALE);
        float sum = 0.f;
        for (int c = 0; c < S_; ++c) {
            float e = expf((float)row[c] * SCALE - m);
            row[c] = (h16)e;
            sum += e;
        }
        const float inv = 1.f / sum;
        for (int c = 0; c < S_; ++c) row[c] = (h16)((float)row[c] * inv);
    }
    __syncthreads();

    // ctx = P[128,128] @ v[128,32]
    v8f accc[2];
    accc[0] = zero; accc[1] = zero;
#pragma unroll
    for (int k0 = 0; k0 < S_; k0 += 32) {
        const h16* ap = sm.sc + (16*wave + l) * S_ + k0 + hi*8;
        v8h lo = *(const v8h*)ap;
        v8h hp = *(const v8h*)(ap + 16);
        v16h a = cat8(lo, hp);
#pragma unroll
        for (int j2 = 0; j2 < 2; ++j2) {
            const h16* bp = sm.vsT + (16*j2 + l) * S_ + k0 + hi*8;
            v16h bb = cat8(*(const v8h*)bp, *(const v8h*)(bp + 16));
            accc[j2] = __builtin_amdgcn_wmma_f32_16x16x32_f16(
                false, a, false, bb, (short)0, accc[j2], false, false);
        }
    }
#pragma unroll
    for (int j2 = 0; j2 < 2; ++j2)
#pragma unroll
        for (int r = 0; r < 8; ++r) {
            const int row = 16*wave + hi*8 + r;
            const int col = 16*j2 + l;
            ctx[((size_t)(bIdx * S_ + row)) * D_ + hIdx * HD_ + col] = (h16)accc[j2][r];
        }
}

// ---------------------------------------------------------------------------
// LayerNorm kernels: one block (256 threads) per row of D=256
// ---------------------------------------------------------------------------
__global__ __launch_bounds__(256) void ln_embed_kernel(
    const float* __restrict__ xt, const float* __restrict__ pos,
    const float* __restrict__ typ, const float* __restrict__ g,
    const float* __restrict__ bb, float* __restrict__ out,
    h16* __restrict__ outh)
{
    __shared__ float red[256];
    const int row = blockIdx.x;
    const int s = row & (S_ - 1);
    const int d = threadIdx.x;
    float v = xt[(size_t)row * D_ + d] + pos[(size_t)s * D_ + d] + typ[d];
    red[d] = v; __syncthreads();
    for (int o = 128; o; o >>= 1) { if (d < o) red[d] += red[d + o]; __syncthreads(); }
    const float mean = red[0] * (1.f / D_);
    __syncthreads();
    const float c = v - mean;
    red[d] = c * c; __syncthreads();
    for (int o = 128; o; o >>= 1) { if (d < o) red[d] += red[d + o]; __syncthreads(); }
    const float var = red[0] * (1.f / D_);
    const float o = c * rsqrtf(var + EPS_) * g[d] + bb[d];
    out[(size_t)row * D_ + d] = o;
    outh[(size_t)row * D_ + d] = (h16)o;
}

__global__ __launch_bounds__(256) void ln_resid_kernel(
    const float* __restrict__ a, const float* __restrict__ dl,
    const float* __restrict__ g, const float* __restrict__ bb,
    float* __restrict__ out, h16* __restrict__ outh)
{
    __shared__ float red[256];
    const int row = blockIdx.x;
    const int d = threadIdx.x;
    const size_t i = (size_t)row * D_ + d;
    float v = a[i] + dl[i];
    red[d] = v; __syncthreads();
    for (int o = 128; o; o >>= 1) { if (d < o) red[d] += red[d + o]; __syncthreads(); }
    const float mean = red[0] * (1.f / D_);
    __syncthreads();
    const float c = v - mean;
    red[d] = c * c; __syncthreads();
    for (int o = 128; o; o >>= 1) { if (d < o) red[d] += red[d + o]; __syncthreads(); }
    const float var = red[0] * (1.f / D_);
    const float o = c * rsqrtf(var + EPS_) * g[d] + bb[d];
    out[i] = o;
    outh[i] = (h16)o;
}

__global__ __launch_bounds__(256) void ln_accum_kernel(
    const float* __restrict__ a, const float* __restrict__ dl,
    const float* __restrict__ g, const float* __restrict__ bb,
    const float* __restrict__ comb, int e, float* __restrict__ z)
{
    __shared__ float red[256];
    const int row = blockIdx.x;
    const int d = threadIdx.x;
    const size_t i = (size_t)row * D_ + d;
    float v = a[i] + dl[i];
    red[d] = v; __syncthreads();
    for (int o = 128; o; o >>= 1) { if (d < o) red[d] += red[d + o]; __syncthreads(); }
    const float mean = red[0] * (1.f / D_);
    __syncthreads();
    const float c = v - mean;
    red[d] = c * c; __syncthreads();
    for (int o = 128; o; o >>= 1) { if (d < o) red[d] += red[d + o]; __syncthreads(); }
    const float var = red[0] * (1.f / D_);
    const float o = c * rsqrtf(var + EPS_) * g[d] + bb[d];
    const int b = row >> 7;   // row / S_
    z[i] += comb[b * E_ + e] * o;
}

// ---------------------------------------------------------------------------
// Router: gate logits -> softmax -> top-2 -> threshold -> comb weights [B,E]
// ---------------------------------------------------------------------------
__global__ __launch_bounds__(256) void router_kernel(
    const float* __restrict__ h_t, const float* __restrict__ e_task,
    const float* __restrict__ e_layout,
    const float* __restrict__ W1, const float* __restrict__ b1,
    const float* __restrict__ W2, const float* __restrict__ b2,
    const float* __restrict__ hW, const float* __restrict__ hb,
    float* __restrict__ gate_probs_out, float* __restrict__ comb)
{
    __shared__ float hid[B_ * HID_];
    __shared__ float lg[B_ * E_];
    __shared__ float pr[B_ * E_];
    __shared__ float tv0[B_], tv1[B_];
    __shared__ int   ti0[B_], ti1[B_];
    __shared__ int   flag[B_];
    __shared__ int   allflag;
    const int t = threadIdx.x;

    for (int idx = t; idx < B_ * HID_; idx += 256) {
        const int b = idx / HID_, j = idx % HID_;
        float s = b1[j];
        const float* ht = h_t + (size_t)b * D_;
        const float* et = e_task + (size_t)b * D_;
        const float* el = e_layout + (size_t)b * D_;
        for (int i = 0; i < D_; ++i) s += ht[i] * W1[(size_t)i * HID_ + j];
        for (int i = 0; i < D_; ++i) s += et[i] * W1[(size_t)(D_ + i) * HID_ + j];
        for (int i = 0; i < D_; ++i) s += el[i] * W1[(size_t)(2*D_ + i) * HID_ + j];
        hid[idx] = fmaxf(s, 0.f);
    }
    __syncthreads();
    if (t < B_ * E_) {
        const int b = t / E_, e = t % E_;
        float s = b2[e] + hb[e];
        for (int j = 0; j < HID_; ++j) s += hid[b * HID_ + j] * W2[j * E_ + e];
        const float* el = e_layout + (size_t)b * D_;
        for (int i = 0; i < D_; ++i) s += el[i] * hW[i * E_ + e];
        lg[t] = s;
    }
    __syncthreads();
    if (t < B_) {
        float m = -1e30f;
        for (int e = 0; e < E_; ++e) m = fmaxf(m, lg[t * E_ + e]);
        float sum = 0.f;
        for (int e = 0; e < E_; ++e) { float ex = expf(lg[t*E_+e] - m); pr[t*E_+e] = ex; sum += ex; }
        const float inv = 1.f / sum;
        float best = -1.f; int bi = 0;
        for (int e = 0; e < E_; ++e) {
            const float pe = pr[t*E_+e] * inv;
            pr[t*E_+e] = pe;
            gate_probs_out[t*E_+e] = pe;
            if (pe > best) { best = pe; bi = e; }
        }
        float sec = -1.f; int si = 0;
        for (int e = 0; e < E_; ++e) {
            if (e == bi) continue;
            const float pe = pr[t*E_+e];
            if (pe > sec) { sec = pe; si = e; }
        }
        tv0[t] = best; ti0[t] = bi; tv1[t] = sec; ti1[t] = si;
        flag[t] = (best > THRESH_) ? 1 : 0;
    }
    __syncthreads();
    if (t == 0) {
        int a = 1;
        for (int b = 0; b < B_; ++b) a &= flag[b];
        allflag = a;
    }
    __syncthreads();
    if (t < B_) {
        for (int e = 0; e < E_; ++e) comb[t * E_ + e] = 0.f;
        comb[t * E_ + ti0[t]] = tv0[t];
        comb[t * E_ + ti1[t]] += allflag ? 0.f : tv1[t];
    }
}

// ---------------------------------------------------------------------------
// Plumbing kernels
// ---------------------------------------------------------------------------
__global__ void build_xt_kernel(const float* __restrict__ h_t,
                                const float* __restrict__ e_task,
                                const float* __restrict__ e_layout,
                                const float* __restrict__ tok,
                                float* __restrict__ xt)
{
    const size_t i = (size_t)blockIdx.x * 256 + threadIdx.x;
    if (i >= (size_t)B_ * S_ * D_) return;
    const int d = (int)(i % D_);
    const int r = (int)((i / D_) % S_);
    const int b = (int)(i / ((size_t)S_ * D_));
    float v;
    if      (r == 0) v = h_t[(size_t)b * D_ + d];
    else if (r == 1) v = e_task[(size_t)b * D_ + d];
    else if (r == 2) v = e_layout[(size_t)b * D_ + d];
    else             v = tok[((size_t)b * N_ + (r - 3)) * D_ + d];
    xt[i] = v;
}

// out[n*K + k] = (f16) in[k*N + n]   (W [K,N] f32 -> W^T [N,K] f16)
__global__ void wtrans_kernel(const float* __restrict__ in, h16* __restrict__ out,
                              int K, int N)
{
    const size_t i = (size_t)blockIdx.x * 256 + threadIdx.x;
    if (i >= (size_t)K * N) return;
    const size_t n = i / K, k = i % K;
    out[i] = (h16)in[k * (size_t)N + n];
}

__global__ void conv_kernel(const float* __restrict__ in, h16* __restrict__ out, size_t n)
{
    const size_t i = (size_t)blockIdx.x * 256 + threadIdx.x;
    if (i < n) out[i] = (h16)in[i];
}

__global__ void zero_kernel(float* __restrict__ p, size_t n)
{
    const size_t i = (size_t)blockIdx.x * 256 + threadIdx.x;
    if (i < n) p[i] = 0.f;
}

// ---------------------------------------------------------------------------
extern "C" void kernel_launch(void* const* d_in, const int* in_sizes, int n_in,
                              void* d_out, int out_size, void* d_ws, size_t ws_size,
                              hipStream_t stream)
{
    (void)in_sizes; (void)n_in; (void)out_size; (void)ws_size;
    const float* h_t       = (const float*)d_in[0];
    const float* e_task    = (const float*)d_in[1];
    const float* e_layout  = (const float*)d_in[2];
    const float* tok       = (const float*)d_in[3];
    const float* rW1       = (const float*)d_in[4];
    const float* rb1       = (const float*)d_in[5];
    const float* rW2       = (const float*)d_in[6];
    const float* rb2       = (const float*)d_in[7];
    const float* hintW     = (const float*)d_in[8];
    const float* hintb     = (const float*)d_in[9];
    const float* pos_emb   = (const float*)d_in[10];
    const float* type_emb  = (const float*)d_in[11];
    const float* ln_emb_g  = (const float*)d_in[12];
    const float* ln_emb_b  = (const float*)d_in[13];
    const float* Wq        = (const float*)d_in[14];
    const float* bq        = (const float*)d_in[15];
    const float* Wk        = (const float*)d_in[16];
    const float* bk        = (const float*)d_in[17];
    const float* Wv        = (const float*)d_in[18];
    const float* bv        = (const float*)d_in[19];
    const float* Wo        = (const float*)d_in[20];
    const float* bo        = (const float*)d_in[21];
    const float* ln_attn_g = (const float*)d_in[22];
    const float* ln_attn_b = (const float*)d_in[23];
    const float* Wf1       = (const float*)d_in[24];
    const float* bf1       = (const float*)d_in[25];
    const float* Wf2       = (const float*)d_in[26];
    const float* bf2       = (const float*)d_in[27];
    const float* ln_ffn_g  = (const float*)d_in[28];
    const float* ln_ffn_b  = (const float*)d_in[29];
    const float* head_W    = (const float*)d_in[30];
    const float* head_b    = (const float*)d_in[31];

    float* logits     = (float*)d_out;
    float* gate_probs = logits + (size_t)B_ * S_ * V_;

    char* ws = (char*)d_ws;
    size_t off = 0;
    auto alloc = [&](size_t bytes) -> char* {
        char* p = ws + off;
        off += (bytes + 255) & ~(size_t)255;
        return p;
    };
    const size_t MT = (size_t)B_ * S_;   // 4096 rows

    float* xt   = (float*)alloc(MT * D_ * 4);
    float* h0   = (float*)alloc(MT * D_ * 4);
    h16*   h0h  = (h16*)  alloc(MT * D_ * 2);
    h16*   qh   = (h16*)  alloc(MT * D_ * 2);
    h16*   kh   = (h16*)  alloc(MT * D_ * 2);
    h16*   vh   = (h16*)  alloc(MT * D_ * 2);
    h16*   ctxh = (h16*)  alloc(MT * D_ * 2);
    float* tmp  = (float*)alloc(MT * D_ * 4);
    float* h1   = (float*)alloc(MT * D_ * 4);
    h16*   h1h  = (h16*)  alloc(MT * D_ * 2);
    h16*   ffh  = (h16*)  alloc(MT * FF_ * 2);
    float* z    = (float*)alloc(MT * D_ * 4);
    h16*   zh   = (h16*)  alloc(MT * D_ * 2);
    float* comb = (float*)alloc((size_t)B_ * E_ * 4);
    h16*   wqT  = (h16*)  alloc((size_t)D_ * D_ * 2);
    h16*   wkT  = (h16*)  alloc((size_t)D_ * D_ * 2);
    h16*   wvT  = (h16*)  alloc((size_t)D_ * D_ * 2);
    h16*   woT  = (h16*)  alloc((size_t)D_ * D_ * 2);
    h16*   wf1T = (h16*)  alloc((size_t)D_ * FF_ * 2);
    h16*   wf2T = (h16*)  alloc((size_t)FF_ * D_ * 2);
    h16*   hWT  = (h16*)  alloc((size_t)D_ * V_ * 2);

    const size_t totXT = MT * D_;
    build_xt_kernel<<<(unsigned)((totXT + 255) / 256), 256, 0, stream>>>(
        h_t, e_task, e_layout, tok, xt);
    router_kernel<<<1, 256, 0, stream>>>(
        h_t, e_task, e_layout, rW1, rb1, rW2, rb2, hintW, hintb, gate_probs, comb);
    zero_kernel<<<(unsigned)((totXT + 255) / 256), 256, 0, stream>>>(z, totXT);
    {
        const size_t hw = (size_t)D_ * V_;
        wtrans_kernel<<<(unsigned)((hw + 255) / 256), 256, 0, stream>>>(head_W, hWT, D_, V_);
    }

    const dim3 blk(256);
    const dim3 gDD(D_ / 128, (unsigned)(MT / 128));   // N=256 GEMMs
    const dim3 gDF(FF_ / 128, (unsigned)(MT / 128));  // N=1024 GEMM
    const dim3 gHV(V_ / 128, (unsigned)(MT / 128));   // head GEMM

    for (int e = 0; e < E_; ++e) {
        const size_t eDD = (size_t)e * D_ * D_;
        const size_t eDF = (size_t)e * D_ * FF_;
        wtrans_kernel<<<(unsigned)(((size_t)D_*D_ + 255)/256), 256, 0, stream>>>(Wq + eDD, wqT, D_, D_);
        wtrans_kernel<<<(unsigned)(((size_t)D_*D_ + 255)/256), 256, 0, stream>>>(Wk + eDD, wkT, D_, D_);
        wtrans_kernel<<<(unsigned)(((size_t)D_*D_ + 255)/256), 256, 0, stream>>>(Wv + eDD, wvT, D_, D_);
        wtrans_kernel<<<(unsigned)(((size_t)D_*D_ + 255)/256), 256, 0, stream>>>(Wo + eDD, woT, D_, D_);
        wtrans_kernel<<<(unsigned)(((size_t)D_*FF_ + 255)/256), 256, 0, stream>>>(Wf1 + eDF, wf1T, D_, FF_);
        wtrans_kernel<<<(unsigned)(((size_t)D_*FF_ + 255)/256), 256, 0, stream>>>(Wf2 + eDF, wf2T, FF_, D_);

        ln_embed_kernel<<<(unsigned)MT, blk, 0, stream>>>(
            xt, pos_emb + (size_t)e * MAXPOS_ * D_, type_emb + (size_t)e * D_,
            ln_emb_g + (size_t)e * D_, ln_emb_b + (size_t)e * D_, h0, h0h);

        gemm_wmma_kernel<<<gDD, blk, 0, stream>>>(h0h, wqT, bq + (size_t)e*D_, nullptr, qh, (int)MT, D_, D_, 0, 0);
        gemm_wmma_kernel<<<gDD, blk, 0, stream>>>(h0h, wkT, bk + (size_t)e*D_, nullptr, kh, (int)MT, D_, D_, 0, 0);
        gemm_wmma_kernel<<<gDD, blk, 0, stream>>>(h0h, wvT, bv + (size_t)e*D_, nullptr, vh, (int)MT, D_, D_, 0, 0);

        attn_kernel<<<B_ * H_, blk, 0, stream>>>(qh, kh, vh, ctxh);

        gemm_wmma_kernel<<<gDD, blk, 0, stream>>>(ctxh, woT, bo + (size_t)e*D_, tmp, nullptr, (int)MT, D_, D_, 0, 0);
        ln_resid_kernel<<<(unsigned)MT, blk, 0, stream>>>(
            h0, tmp, ln_attn_g + (size_t)e*D_, ln_attn_b + (size_t)e*D_, h1, h1h);

        gemm_wmma_kernel<<<gDF, blk, 0, stream>>>(h1h, wf1T, bf1 + (size_t)e*FF_, nullptr, ffh, (int)MT, FF_, D_, 1, 0);
        gemm_wmma_kernel<<<gDD, blk, 0, stream>>>(ffh, wf2T, bf2 + (size_t)e*D_, tmp, nullptr, (int)MT, D_, FF_, 0, 0);
        ln_accum_kernel<<<(unsigned)MT, blk, 0, stream>>>(
            h1, tmp, ln_ffn_g + (size_t)e*D_, ln_ffn_b + (size_t)e*D_, comb, e, z);
    }

    conv_kernel<<<(unsigned)((totXT + 255) / 256), 256, 0, stream>>>(z, zh, totXT);
    // streaming 524MB logits: nontemporal stores to keep L2 for weights
    gemm_wmma_kernel<<<gHV, blk, 0, stream>>>(zh, hWT, head_b, logits, nullptr, (int)MT, V_, D_, 0, 1);
}
